// Attention_47648367182405
// MI455X (gfx1250) — compile-verified, matrix operand order checked by reading, not run on hardware
//
#include <hip/hip_runtime.h>
#include <hip/hip_bf16.h>
#include <math.h>

typedef __attribute__((ext_vector_type(16))) _Float16 v16h;
typedef __attribute__((ext_vector_type(8)))  _Float16 v8h;
typedef __attribute__((ext_vector_type(8)))  float    v8f;

#define HEADS     16
#define DIM_HEAD  64
#define B_SZ      2
#define N_SEQ     2048
#define DIM       1024
#define DIM_INNER (HEADS * DIM_HEAD)   // 1024
#define F_QKV     (3 * DIM_INNER)      // 3072
#define SOFTCAP   50.0f
#define SCALE     0.125f               // 64^-0.5

#ifndef __has_builtin
#define __has_builtin(x) 0
#endif
#if __has_builtin(__builtin_amdgcn_global_load_async_to_lds_b128) && \
    __has_builtin(__builtin_amdgcn_s_wait_asynccnt)
#define HAVE_ASYNC 1
#else
#define HAVE_ASYNC 0
#endif

// The async builtin's params are pointers to 4xi32 vectors in AS1/AS3.
typedef int v4i __attribute__((vector_size(16)));
typedef __attribute__((address_space(1))) v4i* gp4i_t;
typedef __attribute__((address_space(3))) v4i* lp4i_t;

// Copy 16 bytes global -> LDS. Async (ASYNCcnt-tracked) when available.
__device__ inline void async_copy16(const _Float16* g, _Float16* l) {
#if HAVE_ASYNC
  __builtin_amdgcn_global_load_async_to_lds_b128(
      (gp4i_t)(unsigned long long)(size_t)g, (lp4i_t)l, 0, 0);
#else
  *(v8h*)l = *(const v8h*)g;
#endif
}
__device__ inline void wait_async_le4() {
#if HAVE_ASYNC
  __builtin_amdgcn_s_wait_asynccnt(4);
#endif
}
__device__ inline void wait_async_le2() {
#if HAVE_ASYNC
  __builtin_amdgcn_s_wait_asynccnt(2);
#endif
}
__device__ inline void wait_async_le0() {
#if HAVE_ASYNC
  __builtin_amdgcn_s_wait_asynccnt(0);
#endif
}
// Same-wave LDS RAW fence (cross-lane through LDS without a workgroup barrier)
__device__ inline void wave_lds_fence() {
  asm volatile("s_wait_dscnt 0x0" ::: "memory");
}

__device__ inline v8f wmma16x16x32(v16h a, v16h b, v8f c) {
  return __builtin_amdgcn_wmma_f32_16x16x32_f16(
      false, a, false, b, (short)0, c, false, false);
}

// Tile loader (A or B operand) for sources with K contiguous in memory.
// lane<16 : row=lane,    K(h) = h<8 ? h   : h+8   -> two b128 loads
// lane>=16: row=lane-16, K(h) = h<8 ? h+8 : h+16
__device__ inline v16h load_tile_kcontig(const _Float16* p, int ld) {
  int lane = threadIdx.x & 31;
  int row  = lane & 15;
  int koff = (lane >> 4) ? 8 : 0;
  const _Float16* base = p + (size_t)row * ld + koff;
  v8h lo = *(const v8h*)(base);
  v8h hi = *(const v8h*)(base + 16);
  v16h out;
#pragma unroll
  for (int i = 0; i < 8; ++i) { out[i] = lo[i]; out[i + 8] = hi[i]; }
  return out;
}

// B operand from a row-major [K][N] source (K strided): B[k][n] = p[k*ld+n].
__device__ inline v16h load_btile_strided(const _Float16* p, int ld) {
  int lane = threadIdx.x & 31;
  int col  = lane & 15;
  int koff = (lane >> 4) ? 8 : 0;
  v16h out;
#pragma unroll
  for (int h = 0; h < 8; ++h) {
    out[h]     = p[(size_t)(koff + h) * ld + col];
    out[h + 8] = p[(size_t)(koff + 16 + h) * ld + col];
  }
  return out;
}

// ---------------- Kernel 1: RMSNorm + cast to f16 ----------------
__global__ void rmsnorm_cast_kernel(const float* __restrict__ x,
                                    const float* __restrict__ gamma,
                                    _Float16* __restrict__ xn) {
  __shared__ float red[256];
  int row = blockIdx.x;
  int tid = threadIdx.x;
  const float* xr = x + (size_t)row * DIM;
  float s = 0.f;
  for (int i = tid; i < DIM; i += 256) { float v = xr[i]; s += v * v; }
  red[tid] = s;
  __syncthreads();
  for (int w = 128; w > 0; w >>= 1) {
    if (tid < w) red[tid] += red[tid + w];
    __syncthreads();
  }
  float inv = 32.0f / fmaxf(sqrtf(red[0]), 1e-12f);   // sqrt(1024)=32
  _Float16* xo = xn + (size_t)row * DIM;
  for (int i = tid; i < DIM; i += 256)
    xo[i] = (_Float16)(xr[i] * inv * (gamma[i] + 1.0f));
}

// ---------------- Kernel 2: f32 -> f16 cast ----------------
__global__ void cast_f16_kernel(const float* __restrict__ in,
                                _Float16* __restrict__ out, int n) {
  int i = blockIdx.x * blockDim.x + threadIdx.x;
  if (i < n) out[i] = (_Float16)in[i];
}

// ---------------- Kernel 3: fused QKV GEMM ----------------
// 128 threads, 64x64 output tile, async double-buffered B panel in LDS.
__global__ void qkv_gemm_kernel(const _Float16* __restrict__ xn,
                                const _Float16* __restrict__ w,   // [3072,1024]
                                _Float16* __restrict__ q,
                                _Float16* __restrict__ k,
                                _Float16* __restrict__ v) {
  __shared__ __align__(16) _Float16 ldsB[2][64 * 32];
  int col0 = blockIdx.x * 64;    // f
  int row0 = blockIdx.y * 64;    // b*n
  int tid  = threadIdx.x;
  int wave = tid >> 5;
  int arow = row0 + wave * 16;

  auto stage = [&](int kt, int buf) {
    int k0 = kt * 32;
#pragma unroll
    for (int j = tid; j < 256; j += 128) {   // 256 x 16B chunks
      int c = j >> 2, part = j & 3;
      async_copy16(w + (size_t)(col0 + c) * DIM + k0 + part * 8,
                   &ldsB[buf][c * 32 + part * 8]);
    }
  };

  const int KT = DIM / 32;       // 32
  stage(0, 0);
  v8f acc[4] = {};
  for (int kt = 0; kt < KT; ++kt) {
    if (kt + 1 < KT) { stage(kt + 1, (kt + 1) & 1); wait_async_le2(); }
    else             { wait_async_le0(); }
    __syncthreads();
    v16h a = load_tile_kcontig(xn + (size_t)arow * DIM + kt * 32, DIM);
    const _Float16* bb = ldsB[kt & 1];
#pragma unroll
    for (int t = 0; t < 4; ++t)
      acc[t] = wmma16x16x32(a, load_tile_kcontig(bb + t * 16 * 32, 32), acc[t]);
    __syncthreads();
  }

  int lane  = tid & 31;
  int ncol  = lane & 15;
  int mbase = (lane >> 4) ? 8 : 0;
  int which = col0 / DIM_INNER;                 // 0=q 1=k 2=v (const per block)
  int h     = (col0 / DIM_HEAD) % HEADS;
  _Float16* dst = (which == 0) ? q : (which == 1) ? k : v;
  float mul = (which == 0) ? SCALE : 1.0f;
#pragma unroll
  for (int t = 0; t < 4; ++t) {
#pragma unroll
    for (int r = 0; r < 8; ++r) {
      int m  = arow + mbase + r;
      int bx = m / N_SEQ, ns = m % N_SEQ;
      size_t off = (((size_t)bx * HEADS + h) * N_SEQ + ns) * DIM_HEAD
                 + t * 16 + ncol;
      dst[off] = (_Float16)(acc[t][r] * mul);
    }
  }
}

// ---------------- Kernel 4: causal flash attention, tanh softcap ----------
// 128 threads: wave w handles query tile qb*4+w. 32 keys per step (full-K
// P*V WMMAs); K/V pair-tiles async-staged to LDS, shared by all 4 waves.
__global__ void attn_kernel(const _Float16* __restrict__ q,
                            const _Float16* __restrict__ k,
                            const _Float16* __restrict__ v,
                            _Float16* __restrict__ attn_out) {
  __shared__ __align__(16) _Float16 kbuf[2][32 * 64];
  __shared__ __align__(16) _Float16 vbuf[2][32 * 64];
  __shared__ __align__(16) _Float16 pbuf[4][16 * 32];
  const int QB = N_SEQ / 64;                    // 32
  int idx  = blockIdx.x;
  int qb   = idx % QB;
  int h    = (idx / QB) % HEADS;
  int bb   = idx / (QB * HEADS);
  int tid  = threadIdx.x;
  int wave = tid >> 5;
  int lane = tid & 31;
  int qt   = qb * 4 + wave;                     // this wave's 16-query tile
  int np   = qb * 2 + 2;                        // 32-key steps for the block
  size_t head_off = ((size_t)bb * HEADS + h) * (size_t)N_SEQ * DIM_HEAD;
  const _Float16* qp = q + head_off;
  const _Float16* kp = k + head_off;
  const _Float16* vp = v + head_off;

  int ncol  = lane & 15;
  int mbase = (lane >> 4) ? 8 : 0;

  auto stage = [&](int jp, int buf) {   // 32x64 K and V tiles, 4KB contiguous
    const _Float16* ks = kp + (size_t)jp * 32 * DIM_HEAD;
    const _Float16* vs = vp + (size_t)jp * 32 * DIM_HEAD;
#pragma unroll
    for (int j = tid; j < 256; j += 128) {
      async_copy16(ks + j * 8, &kbuf[buf][j * 8]);
      async_copy16(vs + j * 8, &vbuf[buf][j * 8]);
    }
  };

  v16h aq0 = load_tile_kcontig(qp + (size_t)qt * 16 * DIM_HEAD + 0,  DIM_HEAD);
  v16h aq1 = load_tile_kcontig(qp + (size_t)qt * 16 * DIM_HEAD + 32, DIM_HEAD);

  v8f o0 = {}, o1 = {}, o2 = {}, o3 = {};
  float m_run[8], l_run[8];
#pragma unroll
  for (int r = 0; r < 8; ++r) { m_run[r] = -1e30f; l_run[r] = 0.f; }

  stage(0, 0);
  for (int jp = 0; jp < np; ++jp) {
    if (jp + 1 < np) { stage(jp + 1, (jp + 1) & 1); wait_async_le4(); }
    else             { wait_async_le0(); }
    __syncthreads();
    if (jp * 2 <= qt) {                         // at least one unmasked key tile
      const _Float16* kb = kbuf[jp & 1];
      const _Float16* vb = vbuf[jp & 1];
      v8f s0 = {}, s1 = {};
      s0 = wmma16x16x32(aq0, load_tile_kcontig(kb + 0,           DIM_HEAD), s0);
      s0 = wmma16x16x32(aq1, load_tile_kcontig(kb + 32,          DIM_HEAD), s0);
      s1 = wmma16x16x32(aq0, load_tile_kcontig(kb + 16 * 64,     DIM_HEAD), s1);
      s1 = wmma16x16x32(aq1, load_tile_kcontig(kb + 16 * 64 + 32, DIM_HEAD), s1);

      float p0[8], p1[8], alpha[8];
#pragma unroll
      for (int r = 0; r < 8; ++r) {
        float sv0 = SOFTCAP * tanhf(s0[r] * (1.0f / SOFTCAP));
        float sv1 = SOFTCAP * tanhf(s1[r] * (1.0f / SOFTCAP));
        int gr  = qt * 16 + mbase + r;
        int gc0 = jp * 32 + ncol;
        p0[r] = (gc0 > gr)      ? -1e30f : sv0;   // causal mask
        p1[r] = (gc0 + 16 > gr) ? -1e30f : sv1;
      }
#pragma unroll
      for (int r = 0; r < 8; ++r) {
        float rm = fmaxf(p0[r], p1[r]);
        for (int mm = 8; mm >= 1; mm >>= 1) rm = fmaxf(rm, __shfl_xor(rm, mm, 32));
        float mnew = fmaxf(m_run[r], rm);
        alpha[r] = __expf(m_run[r] - mnew);
        p0[r] = __expf(p0[r] - mnew);
        p1[r] = __expf(p1[r] - mnew);
        float rs = p0[r] + p1[r];
        for (int mm = 8; mm >= 1; mm >>= 1) rs += __shfl_xor(rs, mm, 32);
        l_run[r] = l_run[r] * alpha[r] + rs;
        m_run[r] = mnew;
      }
#pragma unroll
      for (int r = 0; r < 8; ++r) {
        o0[r] *= alpha[r]; o1[r] *= alpha[r]; o2[r] *= alpha[r]; o3[r] *= alpha[r];
      }
      // P (two C-tiles) -> per-wave LDS as 16x32 -> full-K A operand
      _Float16* pw = pbuf[wave];
#pragma unroll
      for (int r = 0; r < 8; ++r) {
        pw[(mbase + r) * 32 + ncol]      = (_Float16)p0[r];
        pw[(mbase + r) * 32 + 16 + ncol] = (_Float16)p1[r];
      }
      wave_lds_fence();
      v16h ap = load_tile_kcontig(pw, 32);      // ds_load_b128 x2
      o0 = wmma16x16x32(ap, load_btile_strided(vb + 0,  DIM_HEAD), o0);
      o1 = wmma16x16x32(ap, load_btile_strided(vb + 16, DIM_HEAD), o1);
      o2 = wmma16x16x32(ap, load_btile_strided(vb + 32, DIM_HEAD), o2);
      o3 = wmma16x16x32(ap, load_btile_strided(vb + 48, DIM_HEAD), o3);
    }
    __syncthreads();
  }
#pragma unroll
  for (int r = 0; r < 8; ++r) {
    float inv = 1.0f / l_run[r];
    int m = qt * 16 + mbase + r;
    size_t rowo = ((size_t)bb * N_SEQ + m) * DIM_INNER + (size_t)h * DIM_HEAD;
    attn_out[rowo + 0  + ncol] = (_Float16)(o0[r] * inv);
    attn_out[rowo + 16 + ncol] = (_Float16)(o1[r] * inv);
    attn_out[rowo + 32 + ncol] = (_Float16)(o2[r] * inv);
    attn_out[rowo + 48 + ncol] = (_Float16)(o3[r] * inv);
  }
}

// ---------------- Kernel 5: output projection ----------------
__global__ void out_gemm_kernel(const _Float16* __restrict__ ao,
                                const _Float16* __restrict__ w,  // [1024,1024]
                                float* __restrict__ out) {
  __shared__ __align__(16) _Float16 ldsB[2][64 * 32];
  int col0 = blockIdx.x * 64;
  int row0 = blockIdx.y * 64;
  int tid  = threadIdx.x;
  int wave = tid >> 5;
  int arow = row0 + wave * 16;

  auto stage = [&](int kt, int buf) {
    int k0 = kt * 32;
#pragma unroll
    for (int j = tid; j < 256; j += 128) {
      int c = j >> 2, part = j & 3;
      async_copy16(w + (size_t)(col0 + c) * DIM_INNER + k0 + part * 8,
                   &ldsB[buf][c * 32 + part * 8]);
    }
  };

  const int KT = DIM_INNER / 32;
  stage(0, 0);
  v8f acc[4] = {};
  for (int kt = 0; kt < KT; ++kt) {
    if (kt + 1 < KT) { stage(kt + 1, (kt + 1) & 1); wait_async_le2(); }
    else             { wait_async_le0(); }
    __syncthreads();
    v16h a = load_tile_kcontig(ao + (size_t)arow * DIM_INNER + kt * 32, DIM_INNER);
    const _Float16* bb = ldsB[kt & 1];
#pragma unroll
    for (int t = 0; t < 4; ++t)
      acc[t] = wmma16x16x32(a, load_tile_kcontig(bb + t * 16 * 32, 32), acc[t]);
    __syncthreads();
  }

  int lane  = tid & 31;
  int ncol  = lane & 15;
  int mbase = (lane >> 4) ? 8 : 0;
#pragma unroll
  for (int t = 0; t < 4; ++t)
#pragma unroll
    for (int r = 0; r < 8; ++r)
      out[(size_t)(arow + mbase + r) * DIM + col0 + t * 16 + ncol] = acc[t][r];
}

extern "C" void kernel_launch(void* const* d_in, const int* in_sizes, int n_in,
                              void* d_out, int out_size, void* d_ws, size_t ws_size,
                              hipStream_t stream) {
  const float* x     = (const float*)d_in[0];
  const float* gamma = (const float*)d_in[1];
  const float* w_qkv = (const float*)d_in[2];
  const float* w_out = (const float*)d_in[3];
  float* out = (float*)d_out;

  char* ws = (char*)d_ws;
  const size_t NROWS = (size_t)B_SZ * N_SEQ;                    // 4096
  _Float16* xn   = (_Float16*)(ws + 0);                         // 8 MB
  _Float16* wq16 = (_Float16*)(ws + (8u << 20));                // 6 MB
  _Float16* wo16 = (_Float16*)(ws + (8u << 20) + 6291456u);     // 2 MB
  _Float16* qb   = (_Float16*)(ws + (16u << 20));               // 8 MB
  _Float16* kb   = (_Float16*)(ws + (24u << 20));               // 8 MB
  _Float16* vb   = (_Float16*)(ws + (32u << 20));               // 8 MB
  _Float16* aob  = (_Float16*)(ws + (40u << 20));               // 8 MB

  rmsnorm_cast_kernel<<<dim3((unsigned)NROWS), dim3(256), 0, stream>>>(x, gamma, xn);
  cast_f16_kernel<<<dim3((F_QKV * DIM + 255) / 256), dim3(256), 0, stream>>>(
      w_qkv, wq16, F_QKV * DIM);
  cast_f16_kernel<<<dim3((DIM * DIM_INNER + 255) / 256), dim3(256), 0, stream>>>(
      w_out, wo16, DIM * DIM_INNER);
  qkv_gemm_kernel<<<dim3(F_QKV / 64, (unsigned)(NROWS / 64)), dim3(128), 0, stream>>>(
      xn, wq16, qb, kb, vb);
  attn_kernel<<<dim3(B_SZ * HEADS * (N_SEQ / 64)), dim3(128), 0, stream>>>(
      qb, kb, vb, aob);
  out_gemm_kernel<<<dim3(DIM / 64, (unsigned)(NROWS / 64)), dim3(128), 0, stream>>>(
      aob, wo16, out);
}
